// NASTransformer_55379308314757
// MI455X (gfx1250) — matrix operand order repaired and checked by reading, not compiled, not run on hardware
//
#include <hip/hip_runtime.h>
#include <hip/hip_bf16.h>

// ---------------- model constants ----------------
#define Ec    256
#define Sc    32
#define Bc    64
#define NHc   4
#define HDc   64
#define FFc   1024
#define MAXCc 31
#define INc   68

#define X_TOK   (Bc * Sc)            // 2048
#define P_TOK   (Bc * Sc * Sc)       // 65536
#define TOK_CH  16384                // pair-stream chunk (tokens)
#define NCH     (P_TOK / TOK_CH)     // 4
#define BATCH_CH (TOK_CH / Sc)       // 512

typedef __attribute__((ext_vector_type(16))) __bf16 v16bf;
typedef __attribute__((ext_vector_type(8)))  __bf16 v8bf;
typedef __attribute__((ext_vector_type(8)))  float  v8f;

static __device__ inline __bf16 f2bf(float x) {
  union { float f; unsigned u; } c; c.f = x;
  unsigned r = (c.u + 0x7FFFu + ((c.u >> 16) & 1u)) >> 16;
  unsigned short us = (unsigned short)r;
  return __builtin_bit_cast(__bf16, us);
}

// ---------------- fp32 -> bf16 weight conversion (strided) ----------------
__global__ void k_f32_to_bf16(const float* __restrict__ src, __bf16* __restrict__ dst,
                              int N, int K, int src_stride, int col_off) {
  int idx = blockIdx.x * 256 + threadIdx.x;
  if (idx >= N * K) return;
  int n = idx / K, k = idx - n * K;
  dst[idx] = f2bf(src[(size_t)n * src_stride + col_off + k]);
}

// ---------------- fp32 -> bf16 contiguous activation conversion ----------------
__global__ void k_cvt_flat(const float* __restrict__ src, __bf16* __restrict__ dst,
                           size_t n4) {
  size_t i = (size_t)blockIdx.x * 256 + threadIdx.x;   // float4 id
  if (i >= n4) return;
  float4 f = *(const float4*)(src + i * 4);
  __bf16 o[4] = { f2bf(f.x), f2bf(f.y), f2bf(f.z), f2bf(f.w) };
  *(uint2*)(dst + i * 4) = *(const uint2*)o;
}

// ---------------- WMMA bf16 GEMM: C[M,N] = epi(A[M,K] @ W[N,K]^T + bias) ----------------
// A and W are both bf16 (activations pre-converted by k_cvt_flat).
// EPI: 0 = bias, 1 = bias + exact GELU, 2 = bias + residual add
// Block tile 128x64, per-wave tile 32x32 (acc = 32 VGPRs), BK=64 (two 32-K
// slices -> 8 WMMAs per barrier). Double-buffered LDS; next tile's 6 global
// loads are issued right after the barrier and consumed (ds_store) after the
// WMMAs. Peak VGPR demand ~100, safely under the allocator's 128 cap -> no
// scratch spills in the inner loop.
#define BM 128
#define BN 64
#define BK 64
#define LDP 72   // LDS row pitch (bf16): conflict padding, keeps 16B alignment

template<int EPI>
__global__ __launch_bounds__(256, 1)
void k_gemm(const __bf16* __restrict__ A,
            const __bf16* __restrict__ W,
            const float* __restrict__ bias,
            const float* __restrict__ res,
            float* __restrict__ C,
            int M, int N, int K) {
  __shared__ __bf16 As[2][BM][LDP];
  __shared__ __bf16 Bs[2][BN][LDP];
  const int t    = threadIdx.x;
  const int lane = t & 31;
  const int wave = t >> 5;
  const int wm   = wave & 3;      // 4 waves along M (32 rows each)
  const int wn   = wave >> 2;     // 2 waves along N (32 cols each)
  const int tileM = blockIdx.y * BM;
  const int tileN = blockIdx.x * BN;

  // staging coordinates: A tile 128x64 = 1024 uint4 (4/thread),
  //                      B tile  64x64 =  512 uint4 (2/thread)
  int ar[4], ac[4];
#pragma unroll
  for (int i = 0; i < 4; ++i) {
    int e = t + 256 * i;
    ar[i] = e >> 3;           // row 0..127
    ac[i] = (e & 7) << 3;     // bf16 col 0,8,...,56
  }
  int br[2], bc2[2];
#pragma unroll
  for (int i = 0; i < 2; ++i) {
    int e = t + 256 * i;
    br[i]  = e >> 3;          // row 0..63
    bc2[i] = (e & 7) << 3;
  }

  v8f acc[2][2] = {};
  uint4 fa[4], fb[2];

  const int nk = K / BK;

  // prologue: stage tile 0
#pragma unroll
  for (int i = 0; i < 4; ++i)
    fa[i] = *(const uint4*)(A + (size_t)(tileM + ar[i]) * K + ac[i]);
#pragma unroll
  for (int i = 0; i < 2; ++i)
    fb[i] = *(const uint4*)(W + (size_t)(tileN + br[i]) * K + bc2[i]);
#pragma unroll
  for (int i = 0; i < 4; ++i)
    *((uint4*)&As[0][ar[i]][ac[i]]) = fa[i];
#pragma unroll
  for (int i = 0; i < 2; ++i)
    *((uint4*)&Bs[0][br[i]][bc2[i]]) = fb[i];

  for (int kt = 0; kt < nk; ++kt) {
    const int buf = kt & 1;
    const int k0n = (kt + 1) * BK;

    __syncthreads();   // LDS stage `buf` visible; prior stage reads complete

    // issue next tile's global loads AFTER the barrier: they stay in flight
    // across the fragment ds_loads and the 8 WMMAs below.
    if (kt + 1 < nk) {
#pragma unroll
      for (int i = 0; i < 4; ++i)
        fa[i] = *(const uint4*)(A + (size_t)(tileM + ar[i]) * K + (k0n + ac[i]));
#pragma unroll
      for (int i = 0; i < 2; ++i)
        fb[i] = *(const uint4*)(W + (size_t)(tileN + br[i]) * K + (k0n + bc2[i]));
    }

    // two 32-wide K slices per stage
#pragma unroll
    for (int ks = 0; ks < 2; ++ks) {
      const int kbase = ks * 32;
      // A fragments: lane<16 holds K {0..7,16..23}, lane>=16 holds {8..15,24..31}
      const int kb = kbase + ((lane < 16) ? 0 : 8);
      v16bf af[2];
#pragma unroll
      for (int mt = 0; mt < 2; ++mt) {
        int m = wm * 32 + mt * 16 + (lane & 15);
        union { v16bf v; v8bf h[2]; } u;
        u.h[0] = *(const v8bf*)&As[buf][m][kb];
        u.h[1] = *(const v8bf*)&As[buf][m][16 + kb];
        af[mt] = u.v;
      }
      // B fragment per N tile, consumed immediately by its two WMMAs
      const int kb2 = kbase + ((lane < 16) ? 0 : 8);
#pragma unroll
      for (int nt = 0; nt < 2; ++nt) {
        int n = wn * 32 + nt * 16 + (lane & 15);
        union { v16bf v; v8bf h[2]; } u;
        u.h[0] = *(const v8bf*)&Bs[buf][n][kb2];
        u.h[1] = *(const v8bf*)&Bs[buf][n][kb2 + 16];
        acc[0][nt] = __builtin_amdgcn_wmma_f32_16x16x32_bf16(
            false, af[0], false, u.v, (short)0, acc[0][nt], false, false);
        acc[1][nt] = __builtin_amdgcn_wmma_f32_16x16x32_bf16(
            false, af[1], false, u.v, (short)0, acc[1][nt], false, false);
      }
    }

    // store next tile into the other stage (first load-consumer: the implicit
    // s_wait_loadcnt lands here, after the WMMAs)
    if (kt + 1 < nk) {
      const int nb = buf ^ 1;
#pragma unroll
      for (int i = 0; i < 4; ++i)
        *((uint4*)&As[nb][ar[i]][ac[i]]) = fa[i];
#pragma unroll
      for (int i = 0; i < 2; ++i)
        *((uint4*)&Bs[nb][br[i]][bc2[i]]) = fb[i];
    }
  }

  // epilogue (C/D layout: lanes 0-15 rows r, lanes 16-31 rows 8+r; col = lane&15)
#pragma unroll
  for (int mt = 0; mt < 2; ++mt) {
#pragma unroll
    for (int nt = 0; nt < 2; ++nt) {
      int n = tileN + wn * 32 + nt * 16 + (lane & 15);
      float bv = bias ? bias[n] : 0.0f;
#pragma unroll
      for (int r = 0; r < 8; ++r) {
        int m = tileM + wm * 32 + mt * 16 + ((lane < 16) ? r : (8 + r));
        if (m >= M || n >= N) continue;
        float v = acc[mt][nt][r] + bv;
        if (EPI == 1) v = 0.5f * v * (1.0f + erff(v * 0.7071067811865476f));
        if (EPI == 2) v += res[(size_t)m * N + n];
        C[(size_t)m * N + n] = v;
      }
    }
  }
}

// ---------------- embedding: x[b,s,:] = proj + pos ----------------
__global__ void k_embed(const float* __restrict__ cf, const float* __restrict__ gm,
                        const float* __restrict__ Wc, const float* __restrict__ bc,
                        const float* __restrict__ Wg, const float* __restrict__ bg,
                        const float* __restrict__ pos, float* __restrict__ X) {
  int token = blockIdx.x;                 // 0..2047
  int b = token / Sc, s = token - b * Sc;
  int e = threadIdx.x;                    // 256
  float v;
  if (s == 0) {
    v = gm[b] * Wg[e] + bg[e];
  } else {
    const float* row = cf + ((size_t)b * MAXCc + (s - 1)) * INc;
    const float* w   = Wc + (size_t)e * INc;
    float acc = 0.f;
    for (int k = 0; k < INc; ++k) acc += row[k] * w[k];
    v = acc + bc[e];
  }
  X[(size_t)token * Ec + e] = v + pos[s * Ec + e];
}

// ---------------- c2p broadcast-add: pair[b,i,j,:] (+)= a[b,i,:]+bv[b,j,:]+bias ----------------
template<int ACC>
__global__ void k_c2p(const float* __restrict__ a, const float* __restrict__ bv,
                      const float* __restrict__ bias, float* __restrict__ pair) {
  size_t idx = (size_t)blockIdx.x * 256 + threadIdx.x;  // float4 id
  size_t tot = (size_t)P_TOK * (Ec / 4);
  if (idx >= tot) return;
  int e4 = (int)(idx & 63); size_t r = idx >> 6;
  int j = (int)(r & 31); r >>= 5; int i = (int)(r & 31); int b = (int)(r >> 5);
  float4 av = *(const float4*)(a    + ((size_t)(b * Sc + i)) * Ec + e4 * 4);
  float4 bb = *(const float4*)(bv   + ((size_t)(b * Sc + j)) * Ec + e4 * 4);
  float4 bi = *(const float4*)(bias + e4 * 4);
  float* dst = pair + idx * 4;
  float4 o;
  o.x = av.x + bb.x + bi.x; o.y = av.y + bb.y + bi.y;
  o.z = av.z + bb.z + bi.z; o.w = av.w + bb.w + bi.w;
  if (ACC) { float4 p = *(const float4*)dst; o.x += p.x; o.y += p.y; o.z += p.z; o.w += p.w; }
  *(float4*)dst = o;
}

// ---------------- attention core: one WG per (batch,head), S=32, hd=64 ----------------
__global__ __launch_bounds__(256) void k_attn(const float* __restrict__ qkv,
                                              float* __restrict__ out,
                                              const unsigned char* __restrict__ mask) {
  int bb = blockIdx.x;
  int h  = blockIdx.y;
  __shared__ float q[Sc][HDc], kk[Sc][HDc], vv[Sc][HDc], sc[Sc][Sc];
  const float* base = qkv + (size_t)bb * Sc * (3 * Ec);
  int t = threadIdx.x;
#pragma unroll
  for (int i = 0; i < 8; ++i) {
    int e = t + 256 * i; int row = e >> 6; int d = e & 63;
    const float* rp = base + (size_t)row * (3 * Ec) + h * HDc;
    q[row][d]  = rp[d];
    kk[row][d] = rp[Ec + d];
    vv[row][d] = rp[2 * Ec + d];
  }
  __syncthreads();
#pragma unroll
  for (int i = 0; i < 4; ++i) {
    int e = t + 256 * i; int qi = e >> 5; int kj = e & 31;
    float s = 0.f;
    for (int d = 0; d < HDc; ++d) s += q[qi][d] * kk[kj][d];
    s *= 0.125f;                                  // 1/sqrt(64)
    if (mask) {
      bool valid = (kj == 0) || (mask[bb * MAXCc + kj - 1] != 0);
      if (!valid) s = -1e9f;
    }
    sc[qi][kj] = s;
  }
  __syncthreads();
  if (t < Sc) {
    float mx = -1e30f;
    for (int j = 0; j < Sc; ++j) mx = fmaxf(mx, sc[t][j]);
    float sum = 0.f;
    for (int j = 0; j < Sc; ++j) { float ev = __expf(sc[t][j] - mx); sc[t][j] = ev; sum += ev; }
    float inv = 1.0f / sum;
    for (int j = 0; j < Sc; ++j) sc[t][j] *= inv;
  }
  __syncthreads();
#pragma unroll
  for (int i = 0; i < 8; ++i) {
    int e = t + 256 * i; int row = e >> 6; int d = e & 63;
    float o = 0.f;
    for (int j = 0; j < Sc; ++j) o += sc[row][j] * vv[j][d];
    out[((size_t)bb * Sc + row) * Ec + h * HDc + d] = o;
  }
}

// ---------------- LayerNorm (wave per 256-wide row): out = LN(x(+res))*g+b ----------------
__global__ void k_ln(const float* __restrict__ x, long xstride,
                     const float* __restrict__ res,
                     const float* __restrict__ g, const float* __restrict__ bta,
                     float* __restrict__ out, int rows) {
  int lane = threadIdx.x & 31;
  int w    = threadIdx.x >> 5;
  int row  = blockIdx.x * 8 + w;
  if (row >= rows) return;
  const float* xr = x + (size_t)row * xstride;
  float vals[8];
  float s = 0.f;
#pragma unroll
  for (int i = 0; i < 8; ++i) {
    int e = i * 32 + lane;
    float v = xr[e];
    if (res) v += res[(size_t)row * Ec + e];
    vals[i] = v; s += v;
  }
  for (int o = 16; o > 0; o >>= 1) s += __shfl_xor(s, o, 32);
  float mean = s * (1.0f / Ec);
  float vs = 0.f;
#pragma unroll
  for (int i = 0; i < 8; ++i) { float d = vals[i] - mean; vs += d * d; }
  for (int o = 16; o > 0; o >>= 1) vs += __shfl_xor(vs, o, 32);
  float inv = rsqrtf(vs * (1.0f / Ec) + 1e-5f);
#pragma unroll
  for (int i = 0; i < 8; ++i) {
    int e = i * 32 + lane;
    out[(size_t)row * Ec + e] = (vals[i] - mean) * inv * g[e] + bta[e];
  }
}

// ---------------- mean over axis 2: out[b,i,:] = mean_j pair[b,i,j,:] ----------------
__global__ void k_mean_j(const float* __restrict__ pair, float* __restrict__ out) {
  size_t idx = (size_t)blockIdx.x * 256 + threadIdx.x;   // over B*S*E
  if (idx >= (size_t)X_TOK * Ec) return;
  int e = (int)(idx & (Ec - 1)); size_t r = idx >> 8;    // r = b*S+i
  const float* p = pair + r * (size_t)Sc * Ec + e;
  float s = 0.f;
  for (int j = 0; j < Sc; ++j) s += p[(size_t)j * Ec];
  out[idx] = s * (1.0f / Sc);
}

// ---------------- pair transpose: out[b,j,i,:] = in[b,i,j,:] ----------------
__global__ void k_transpose(const float* __restrict__ in, float* __restrict__ out) {
  size_t idx = (size_t)blockIdx.x * 256 + threadIdx.x;   // float4 id
  size_t tot = (size_t)P_TOK * (Ec / 4);
  if (idx >= tot) return;
  int e4 = (int)(idx & 63); size_t r = idx >> 6;
  int j = (int)(r & 31); r >>= 5; int i = (int)(r & 31); int b = (int)(r >> 5);
  float4 v = *(const float4*)(in  + (((size_t)(b * Sc + i) * Sc + j) * Ec) + e4 * 4);
  *(float4*)(out + (((size_t)(b * Sc + j) * Sc + i) * Ec) + e4 * 4) = v;
}

// ---------------- head: pooler tanh and prediction ----------------
__global__ void k_pool(const float* __restrict__ pin, const float* __restrict__ W,
                       const float* __restrict__ b, float* __restrict__ pout) {
  int bi = blockIdx.x; int n = threadIdx.x;
  const float* xr = pin + (size_t)bi * Ec;
  const float* wr = W + (size_t)n * Ec;
  float s = 0.f;
  for (int k = 0; k < Ec; ++k) s += xr[k] * wr[k];
  pout[(size_t)bi * Ec + n] = tanhf(s + b[n]);
}

__global__ void k_pred(const float* __restrict__ pin, const float* __restrict__ W,
                       const float* __restrict__ b, float* __restrict__ out) {
  int bi = blockIdx.x; int o = threadIdx.x;
  if (o >= 3) return;
  const float* xr = pin + (size_t)bi * Ec;
  const float* wr = W + (size_t)o * Ec;
  float s = 0.f;
  for (int k = 0; k < Ec; ++k) s += xr[k] * wr[k];
  out[(size_t)bi * 3 + o] = s + b[o];
}

// =====================================================================
extern "C" void kernel_launch(void* const* d_in, const int* in_sizes, int n_in,
                              void* d_out, int out_size, void* d_ws, size_t ws_size,
                              hipStream_t stream) {
  (void)in_sizes; (void)n_in; (void)out_size; (void)ws_size;
  auto F = [&](int i) -> const float* { return (const float*)d_in[i]; };
  const unsigned char* mask = (const unsigned char*)d_in[2];

  // ---- stage 1: convert all GEMM weights fp32 -> bf16 into workspace ----
  __bf16* wbase = (__bf16*)d_ws;
  size_t woff = 0;
  auto conv = [&](const float* src, int N, int K, int stride, int coloff) -> const __bf16* {
    __bf16* dst = wbase + woff; woff += (size_t)N * K;
    int tot = N * K;
    k_f32_to_bf16<<<(tot + 255) / 256, 256, 0, stream>>>(src, dst, N, K, stride, coloff);
    return dst;
  };

  const __bf16* c2p_wi = conv(F(8),  Ec, Ec, 2 * Ec, 0);
  const __bf16* c2p_wj = conv(F(8),  Ec, Ec, 2 * Ec, Ec);
  const __bf16* p2c_w  = conv(F(10), Ec, Ec, Ec, 0);
  struct CW { const __bf16 *qkv, *wo, *w1, *w2; } cw[3];
  struct PW { const __bf16 *rqkv, *rwo, *cqkv, *cwo, *w1, *w2; } pw[3];
  for (int l = 0; l < 3; ++l) {
    int cb = 12 + 12 * l;
    cw[l].qkv = conv(F(cb + 0), 3 * Ec, Ec, Ec, 0);
    cw[l].wo  = conv(F(cb + 2), Ec, Ec, Ec, 0);
    cw[l].w1  = conv(F(cb + 6), FFc, Ec, Ec, 0);
    cw[l].w2  = conv(F(cb + 8), Ec, FFc, FFc, 0);
    int pb = 48 + 18 * l;
    pw[l].rqkv = conv(F(pb + 0), 3 * Ec, Ec, Ec, 0);
    pw[l].rwo  = conv(F(pb + 2), Ec, Ec, Ec, 0);
    pw[l].cqkv = conv(F(pb + 4), 3 * Ec, Ec, Ec, 0);
    pw[l].cwo  = conv(F(pb + 6), Ec, Ec, Ec, 0);
    pw[l].w1   = conv(F(pb + 14), FFc, Ec, Ec, 0);
    pw[l].w2   = conv(F(pb + 16), Ec, FFc, FFc, 0);
  }

  // ---- workspace layout ----
  size_t wbytes = (woff * sizeof(__bf16) + 255) & ~(size_t)255;
  float* fbase = (float*)((char*)d_ws + wbytes);
  size_t fo = 0;
  auto alloc = [&](size_t n) { float* p = fbase + fo; fo += n; return p; };
  // bf16 scratch for the pre-converted GEMM A operand (max M*K = 16384*1024)
  __bf16* ABH = (__bf16*)alloc(((size_t)TOK_CH * FFc) / 2);
  float* X0   = alloc((size_t)X_TOK * Ec);
  float* X1   = alloc((size_t)X_TOK * Ec);
  float* CA   = alloc((size_t)X_TOK * Ec);
  float* CB   = alloc((size_t)X_TOK * Ec);
  float* CQKV = alloc((size_t)X_TOK * 3 * Ec);
  float* PAIR = alloc((size_t)P_TOK * Ec);
  float* PBUF = alloc((size_t)P_TOK * Ec);
  float* QKVC = alloc((size_t)TOK_CH * 3 * Ec);
  float* AOUT = alloc((size_t)TOK_CH * Ec);
  float* FFH  = alloc((size_t)TOK_CH * FFc);   // also reused as col-proj scratch
  float* POOL  = alloc((size_t)Bc * Ec);
  float* POOL2 = alloc((size_t)Bc * Ec);

  auto gemm = [&](int epi, const float* A, const __bf16* W, const float* bias,
                  const float* res, float* C, int M, int N, int K) {
    size_t n4 = ((size_t)M * K) / 4;
    k_cvt_flat<<<(unsigned)((n4 + 255) / 256), 256, 0, stream>>>(A, ABH, n4);
    dim3 grid(N / BN, M / BM);
    switch (epi) {
      case 0: k_gemm<0><<<grid, 256, 0, stream>>>(ABH, W, bias, res, C, M, N, K); break;
      case 1: k_gemm<1><<<grid, 256, 0, stream>>>(ABH, W, bias, res, C, M, N, K); break;
      default: k_gemm<2><<<grid, 256, 0, stream>>>(ABH, W, bias, res, C, M, N, K); break;
    }
  };

  const int P4  = (int)((size_t)P_TOK * (Ec / 4) / 256);  // 16384 blocks
  float* X = X0; float* XT = X1;

  // ---- embed ----
  k_embed<<<X_TOK, 256, 0, stream>>>(F(0), F(1), F(3), F(4), F(5), F(6), F(7), X);

  // ---- initial pair = c2p(x) ----
  gemm(0, X, c2p_wi, nullptr, nullptr, CA, X_TOK, Ec, Ec);
  gemm(0, X, c2p_wj, nullptr, nullptr, CB, X_TOK, Ec, Ec);
  k_c2p<0><<<P4, 256, 0, stream>>>(CA, CB, F(9), PAIR);

  for (int l = 0; l < 3; ++l) {
    int cb = 12 + 12 * l;
    int pb = 48 + 18 * l;

    // ===== component layer =====
    gemm(0, X, cw[l].qkv, F(cb + 1), nullptr, CQKV, X_TOK, 3 * Ec, Ec);
    k_attn<<<dim3(Bc, NHc), 256, 0, stream>>>(CQKV, AOUT, mask);
    gemm(2, AOUT, cw[l].wo, F(cb + 3), X, XT, X_TOK, Ec, Ec);
    k_ln<<<X_TOK / 8, 256, 0, stream>>>(XT, Ec, nullptr, F(cb + 4), F(cb + 5), X, X_TOK);
    gemm(1, X, cw[l].w1, F(cb + 7), nullptr, FFH, X_TOK, FFc, Ec);
    gemm(2, FFH, cw[l].w2, F(cb + 9), X, XT, X_TOK, Ec, FFc);
    k_ln<<<X_TOK / 8, 256, 0, stream>>>(XT, Ec, nullptr, F(cb + 10), F(cb + 11), X, X_TOK);

    // ===== pair accumulation (layers > 0) =====
    if (l > 0) {
      gemm(0, X, c2p_wi, nullptr, nullptr, CA, X_TOK, Ec, Ec);
      gemm(0, X, c2p_wj, nullptr, nullptr, CB, X_TOK, Ec, Ec);
      k_c2p<1><<<P4, 256, 0, stream>>>(CA, CB, F(9), PAIR);
    }

    // ===== pair layer: row attention =====
    for (int c = 0; c < NCH; ++c) {
      float* Pc = PAIR + (size_t)c * TOK_CH * Ec;
      float* Oc = PBUF + (size_t)c * TOK_CH * Ec;
      gemm(0, Pc, pw[l].rqkv, F(pb + 1), nullptr, QKVC, TOK_CH, 3 * Ec, Ec);
      k_attn<<<dim3(BATCH_CH, NHc), 256, 0, stream>>>(QKVC, AOUT, nullptr);
      gemm(2, AOUT, pw[l].rwo, F(pb + 3), Pc, Oc, TOK_CH, Ec, Ec);
    }
    k_ln<<<P_TOK / 8, 256, 0, stream>>>(PBUF, Ec, nullptr, F(pb + 8), F(pb + 9), PAIR, P_TOK);

    // ===== pair layer: column attention (via transpose) =====
    k_transpose<<<P4, 256, 0, stream>>>(PAIR, PBUF);
    for (int c = 0; c < NCH; ++c) {
      float* Pc = PBUF + (size_t)c * TOK_CH * Ec;
      float* Oc = FFH  + (size_t)c * TOK_CH * Ec;   // col-proj (transposed layout)
      gemm(0, Pc, pw[l].cqkv, F(pb + 5), nullptr, QKVC, TOK_CH, 3 * Ec, Ec);
      k_attn<<<dim3(BATCH_CH, NHc), 256, 0, stream>>>(QKVC, AOUT, nullptr);
      gemm(0, AOUT, pw[l].cwo, F(pb + 7), nullptr, Oc, TOK_CH, Ec, Ec);
    }
    k_transpose<<<P4, 256, 0, stream>>>(FFH, PBUF);   // back to [b,i,j,:]
    k_ln<<<P_TOK / 8, 256, 0, stream>>>(PBUF, Ec, PAIR, F(pb + 10), F(pb + 11), PAIR, P_TOK);

    // ===== pair layer: FF =====
    for (int c = 0; c < NCH; ++c) {
      float* Pc = PAIR + (size_t)c * TOK_CH * Ec;
      float* Oc = PBUF + (size_t)c * TOK_CH * Ec;
      gemm(1, Pc, pw[l].w1, F(pb + 15), nullptr, FFH, TOK_CH, FFc, Ec);
      gemm(2, FFH, pw[l].w2, F(pb + 17), Pc, Oc, TOK_CH, Ec, FFc);
    }
    k_ln<<<P_TOK / 8, 256, 0, stream>>>(PBUF, Ec, nullptr, F(pb + 12), F(pb + 13), PAIR, P_TOK);

    // ===== x = x + p2c(pair) =====
    k_mean_j<<<(X_TOK * Ec) / 256, 256, 0, stream>>>(PAIR, CA);
    gemm(2, CA, p2c_w, F(11), X, XT, X_TOK, Ec, Ec);
    { float* tmp = X; X = XT; XT = tmp; }
  }

  // ---- head ----
  k_ln<<<Bc / 8, 256, 0, stream>>>(X, (long)Sc * Ec, nullptr, F(102), F(103), POOL, Bc);
  k_pool<<<Bc, Ec, 0, stream>>>(POOL, F(104), F(105), POOL2);
  k_pred<<<Bc, 32, 0, stream>>>(POOL2, F(106), F(107), (float*)d_out);
}
